// SerialBlock_adapt_M_37142877176327
// MI455X (gfx1250) — matrix operand order repaired and checked by reading, not compiled, not run on hardware
//
#include <hip/hip_runtime.h>
#include <hip/hip_bf16.h>
#include <cmath>

// ---------------- types ----------------
typedef __bf16 bf16_t;
typedef __attribute__((ext_vector_type(16))) __bf16 v16bf;
typedef __attribute__((ext_vector_type(8)))  __bf16 v8bf;
typedef __attribute__((ext_vector_type(8)))  float  v8f;

// async-to-LDS builtin takes v4i* in AS1 (global) and AS3 (LDS)
typedef int v4i_vs __attribute__((vector_size(16)));
typedef __attribute__((address_space(1))) v4i_vs* gp128;
typedef __attribute__((address_space(3))) v4i_vs* lp128;

#define Bn   8
#define Hh   56
#define Ww   56
#define Nn   3136          // 56*56
#define Cc   256
#define NH   8
#define CH   32
#define MM   (Bn*Nn)       // 25088 rows
#define HID  1024
#define SCALE 0.17677669529663687f  // 32^-0.5

#if defined(__gfx1250__) && __has_builtin(__builtin_amdgcn_global_load_async_to_lds_b128)
#define HAVE_ASYNC_LDS 1
#else
#define HAVE_ASYNC_LDS 0
#endif

// =====================================================================
// Weight convert + transpose: Wt[n*K + k] = bf16(W[k*N + n])
// =====================================================================
__global__ void wconv_kernel(const float* __restrict__ W, bf16_t* __restrict__ Wt,
                             int K, int N) {
  long i = (long)blockIdx.x * 256 + threadIdx.x;
  if (i >= (long)K * N) return;
  int n = (int)(i / K);
  int k = (int)(i % K);
  Wt[i] = (bf16_t)W[(long)k * N + n];
}

// =====================================================================
// CPE: depthwise 3x3 'same' conv on [B,C,H,W] view of x[B,N,C] + residual
// =====================================================================
__global__ void cpe_kernel(const float* __restrict__ x, const float* __restrict__ w,
                           const float* __restrict__ bias, float* __restrict__ x1) {
  int c = threadIdx.x;
  int n = blockIdx.x % Nn;
  int b = blockIdx.x / Nn;
  int y = n / Ww, xx = n % Ww;
  const float* xb = x + (size_t)b * Nn * Cc;
  float acc = bias[c];
  #pragma unroll
  for (int dy = -1; dy <= 1; ++dy) {
    int yy = y + dy;
    if (yy < 0 || yy >= Hh) continue;
    #pragma unroll
    for (int dx = -1; dx <= 1; ++dx) {
      int xc = xx + dx;
      if (xc < 0 || xc >= Ww) continue;
      acc += w[c * 9 + (dy + 1) * 3 + (dx + 1)] * xb[(size_t)(yy * Ww + xc) * Cc + c];
    }
  }
  x1[(size_t)blockIdx.x * Cc + c] = acc + xb[(size_t)n * Cc + c];
}

// =====================================================================
// LayerNorm over C=256 (domain-indexed gamma/beta), bf16 output
// =====================================================================
__global__ void ln_kernel(const float* __restrict__ xin, const float* __restrict__ g,
                          const float* __restrict__ beta, const int* __restrict__ dptr,
                          bf16_t* __restrict__ out) {
  int c = threadIdx.x;
  size_t row = blockIdx.x;
  float v = xin[row * Cc + c];
  float s1 = v, s2 = v * v;
  #pragma unroll
  for (int o = 16; o > 0; o >>= 1) {
    s1 += __shfl_xor(s1, o, 32);
    s2 += __shfl_xor(s2, o, 32);
  }
  __shared__ float r1[8], r2[8];
  int wid = c >> 5;
  if ((c & 31) == 0) { r1[wid] = s1; r2[wid] = s2; }
  __syncthreads();
  float t1 = 0.f, t2 = 0.f;
  #pragma unroll
  for (int i = 0; i < 8; ++i) { t1 += r1[i]; t2 += r2[i]; }
  float mean = t1 * (1.f / Cc);
  float var  = t2 * (1.f / Cc) - mean * mean;
  float inv  = rsqrtf(var + 1e-5f);
  int d = dptr[0];
  float yv = (v - mean) * inv * g[d * Cc + c] + beta[d * Cc + c];
  out[row * Cc + c] = (bf16_t)yv;
}

// =====================================================================
// WMMA bf16 GEMM: D = A[M,K] x Bt[N,K]^T  (+bias)(+resid)(GELU)
// Block = 4 waves = 128x64 tile. Each wave: 32x64 (2 M-frags x 4 N-frags,
// 8 WMMAs / K-step). B tile (64x32 bf16 = 4KB) staged in LDS per K-step
// via GLOBAL_LOAD_ASYNC_TO_LDS_B128 (ASYNCcnt) when available.
// Fragment layouts per CDNA5 ISA 7.12.2.
// =====================================================================
#define EPI_PLAIN       0   // f32 out
#define EPI_BIAS_RESID  1   // f32 out = d + bias[col] + resid[row,col]
#define EPI_BIAS_GELU   2   // bf16 out = gelu(d + bias[col])

template <int EPI>
__global__ void gemm_bf16_kernel(const bf16_t* __restrict__ A,
                                 const bf16_t* __restrict__ Bt,
                                 const float* __restrict__ bias,
                                 const float* __restrict__ resid,
                                 float* __restrict__ Of,
                                 bf16_t* __restrict__ Ob,
                                 int M, int N, int K) {
  __shared__ bf16_t sB[64 * 32];          // [col][k'] bf16, 4KB

  int tid  = threadIdx.x;                 // 0..127
  int wave = tid >> 5;
  int lane = tid & 31;
  int hf   = lane >> 4;                   // half-wave select
  int r    = lane & 15;

  int m0 = blockIdx.x * 128 + wave * 32;  // this wave's 32 rows
  int n0 = blockIdx.y * 64;               // block's 64 cols

  const bf16_t* arow0 = A + (size_t)(m0 + r) * K;        // M-frag 0
  const bf16_t* arow1 = A + (size_t)(m0 + 16 + r) * K;   // M-frag 1

  // staging assignment: 256 16B-chunks, 2 per thread; chunk c -> col=c>>2, part=c&3
  int sc0 = tid, sc1 = tid + 128;

  v8f acc[2][4] = {};

  for (int kb = 0; kb < K; kb += 32) {
    // ---- stage B tile: cols n0..n0+63, K window kb..kb+31 ----
    #pragma unroll
    for (int u = 0; u < 2; ++u) {
      int c    = u ? sc1 : sc0;
      int col  = c >> 2, part = c & 3;
      const bf16_t* g = Bt + (size_t)(n0 + col) * K + kb + part * 8;
      bf16_t*       l = &sB[col * 32 + part * 8];
#if HAVE_ASYNC_LDS
      __builtin_amdgcn_global_load_async_to_lds_b128(
          (gp128)(uintptr_t)g, (lp128)(uintptr_t)l, 0, 0);
#else
      *(v8bf*)l = *(const v8bf*)g;
#endif
    }
#if HAVE_ASYNC_LDS
#if __has_builtin(__builtin_amdgcn_s_wait_asynccnt)
    __builtin_amdgcn_s_wait_asynccnt(0);
#else
    asm volatile("s_wait_asynccnt 0" ::: "memory");
#endif
#endif
    __syncthreads();

    // ---- A fragments (direct global), 16x32 bf16 layout ----
    union { v16bf v; v8bf h[2]; } a0, a1;
    a0.h[0] = *(const v8bf*)(arow0 + kb + hf * 8);
    a0.h[1] = *(const v8bf*)(arow0 + kb + 16 + hf * 8);
    a1.h[0] = *(const v8bf*)(arow1 + kb + hf * 8);
    a1.h[1] = *(const v8bf*)(arow1 + kb + 16 + hf * 8);

    // ---- B fragments from LDS + 8 WMMAs ----
    #pragma unroll
    for (int f = 0; f < 4; ++f) {
      union { v16bf v; v8bf h[2]; } b;
      const bf16_t* l = &sB[(f * 16 + r) * 32 + hf * 8];
      b.h[0] = *(const v8bf*)l;
      b.h[1] = *(const v8bf*)(l + 16);
      acc[0][f] = __builtin_amdgcn_wmma_f32_16x16x32_bf16(
          false, a0.v, false, b.v, (short)0, acc[0][f], false, false);
      acc[1][f] = __builtin_amdgcn_wmma_f32_16x16x32_bf16(
          false, a1.v, false, b.v, (short)0, acc[1][f], false, false);
    }
    __syncthreads();
  }

  // ---- epilogue: C/D 16x16 f32 layout: VGPR j -> row +hf*8+j, col +r ----
  #pragma unroll
  for (int t = 0; t < 2; ++t) {
    #pragma unroll
    for (int f = 0; f < 4; ++f) {
      #pragma unroll
      for (int j = 0; j < 8; ++j) {
        int row = m0 + t * 16 + hf * 8 + j;
        int col = n0 + f * 16 + r;
        size_t idx = (size_t)row * N + col;
        float v = acc[t][f][j];
        if constexpr (EPI == EPI_PLAIN) {
          Of[idx] = v;
        } else if constexpr (EPI == EPI_BIAS_RESID) {
          Of[idx] = v + bias[col] + resid[idx];
        } else { // EPI_BIAS_GELU
          v += bias[col];
          float ge = 0.5f * v * (1.f + erff(v * 0.7071067811865476f));
          Ob[idx] = (bf16_t)ge;
        }
      }
    }
  }
}

// =====================================================================
// kv einsum with token softmax on k:
// kv[b,h,ck,cv] = sum_n softmax_n(k)[n,ck] * v[n,cv];  one block per (b,h)
// =====================================================================
__global__ void kv_kernel(const float* __restrict__ qkv, float* __restrict__ kvout) {
  int bh = blockIdx.x;
  int b = bh >> 3, hh = bh & 7;
  const float* kb = qkv + (size_t)b * Nn * 768 + 256 + hh * CH;
  const float* vb = qkv + (size_t)b * Nn * 768 + 512 + hh * CH;
  int t = threadIdx.x;

  // phase 1: per-column max over N
  __shared__ float tmp[256];
  __shared__ float mx[CH];
  {
    int col = t & 31, grp = t >> 5;
    float pm = -3.4e38f;
    for (int n = grp; n < Nn; n += 8) pm = fmaxf(pm, kb[(size_t)n * 768 + col]);
    tmp[t] = pm;
    __syncthreads();
    if (t < 32) {
      float m = tmp[t];
      #pragma unroll
      for (int g = 1; g < 8; ++g) m = fmaxf(m, tmp[g * 32 + t]);
      mx[t] = m;
    }
    __syncthreads();
  }

  // phase 2: accumulate exp(k-max)*v and sum(exp)
  float acc[4] = {0.f, 0.f, 0.f, 0.f};
  float se = 0.f;
  int ck = t >> 3, cvb = (t & 7) * 4;
  __shared__ float sk[4][CH], sv[4][CH];
  for (int n0 = 0; n0 < Nn; n0 += 4) {
    int tok = t >> 6, idx = t & 63;
    int n = n0 + tok;
    if (idx < 32) sk[tok][idx]      = kb[(size_t)n * 768 + idx];
    else          sv[tok][idx - 32] = vb[(size_t)n * 768 + (idx - 32)];
    __syncthreads();
    #pragma unroll
    for (int tt = 0; tt < 4; ++tt) {
      float e = __expf(sk[tt][ck] - mx[ck]);
      if (cvb == 0) se += e;
      #pragma unroll
      for (int j = 0; j < 4; ++j) acc[j] += e * sv[tt][cvb + j];
    }
    __syncthreads();
  }
  __shared__ float ssum[CH];
  if (cvb == 0) ssum[ck] = se;
  __syncthreads();
  float inv = 1.f / ssum[ck];
  float* o = kvout + (size_t)bh * 1024 + ck * 32 + cvb;
  #pragma unroll
  for (int j = 0; j < 4; ++j) o[j] = acc[j] * inv;
}

// =====================================================================
// CRPE: channel-grouped depthwise conv on v image {c<64:3x3, <160:5x5, else 7x7}
// =====================================================================
__global__ void crpe_kernel(const float* __restrict__ qkv,
                            const float* __restrict__ w3, const float* __restrict__ b3,
                            const float* __restrict__ w5, const float* __restrict__ b5,
                            const float* __restrict__ w7, const float* __restrict__ b7,
                            float* __restrict__ conv_v) {
  int c = threadIdx.x;
  int n = blockIdx.x % Nn;
  int b = blockIdx.x / Nn;
  int y = n / Ww, xx = n % Ww;
  const float* vb = qkv + (size_t)b * Nn * 768 + 512;
  int ks; const float* w; float acc;
  if (c < 64)        { ks = 3; w = w3 + c * 9;          acc = b3[c]; }
  else if (c < 160)  { ks = 5; w = w5 + (c - 64) * 25;  acc = b5[c - 64]; }
  else               { ks = 7; w = w7 + (c - 160) * 49; acc = b7[c - 160]; }
  int hk = ks >> 1;
  for (int dy = -hk; dy <= hk; ++dy) {
    int yy = y + dy;
    if (yy < 0 || yy >= Hh) continue;
    for (int dx = -hk; dx <= hk; ++dx) {
      int xc = xx + dx;
      if (xc < 0 || xc >= Ww) continue;
      acc += w[(dy + hk) * ks + (dx + hk)] * vb[(size_t)(yy * Ww + xc) * 768 + c];
    }
  }
  conv_v[(size_t)blockIdx.x * Cc + c] = acc;
}

// =====================================================================
// Domain gating: da = softmax_heads( relu(dl@w1+b1)@w2 + b2 )   [B,256]
// =====================================================================
__global__ void da_kernel(const float* __restrict__ dl,
                          const float* __restrict__ w1, const float* __restrict__ b1,
                          const float* __restrict__ w2, const float* __restrict__ b2,
                          float* __restrict__ da) {
  __shared__ float hid[Bn][128];
  __shared__ float o[Bn][Cc];
  int t = threadIdx.x;
  for (int i = t; i < Bn * 128; i += 256) {
    int b = i >> 7, j = i & 127;
    float a = b1[j];
    #pragma unroll
    for (int k = 0; k < 4; ++k) a += dl[b * 4 + k] * w1[k * 128 + j];
    hid[b][j] = fmaxf(a, 0.f);
  }
  __syncthreads();
  for (int i = t; i < Bn * Cc; i += 256) {
    int b = i >> 8, j = i & 255;
    float a = b2[j];
    for (int k = 0; k < 128; ++k) a += hid[b][k] * w2[k * Cc + j];
    o[b][j] = a;
  }
  __syncthreads();
  int b = t >> 5, ch = t & 31;
  float m = -3.4e38f;
  #pragma unroll
  for (int hh = 0; hh < NH; ++hh) m = fmaxf(m, o[b][hh * CH + ch]);
  float e[NH]; float s = 0.f;
  #pragma unroll
  for (int hh = 0; hh < NH; ++hh) { e[hh] = __expf(o[b][hh * CH + ch] - m); s += e[hh]; }
  float inv = 1.f / s;
  #pragma unroll
  for (int hh = 0; hh < NH; ++hh) da[(size_t)b * Cc + hh * CH + ch] = e[hh] * inv;
}

// =====================================================================
// Fuse: attn[b,n,c] = da[b,c]*(scale * q@kv + q*conv_v)  -> bf16
// =====================================================================
__global__ void fuse_kernel(const float* __restrict__ qkv, const float* __restrict__ kv,
                            const float* __restrict__ conv_v, const float* __restrict__ da,
                            bf16_t* __restrict__ attn_bf) {
  int c = threadIdx.x;
  int n = blockIdx.x % Nn;
  int b = blockIdx.x / Nn;
  __shared__ float qrow[Cc];
  const float* qr = qkv + (size_t)(b * Nn + n) * 768;
  qrow[c] = qr[c];
  __syncthreads();
  int hh = c >> 5, cv = c & 31;
  const float* kvb = kv + (size_t)(b * NH + hh) * 1024;
  float fa = 0.f;
  #pragma unroll
  for (int ck = 0; ck < CH; ++ck) fa += qrow[hh * CH + ck] * kvb[ck * CH + cv];
  float val = da[(size_t)b * Cc + c] *
              (SCALE * fa + qrow[c] * conv_v[(size_t)blockIdx.x * Cc + c]);
  attn_bf[(size_t)blockIdx.x * Cc + c] = (bf16_t)val;
}

// =====================================================================
// launcher
// =====================================================================
extern "C" void kernel_launch(void* const* d_in, const int* in_sizes, int n_in,
                              void* d_out, int out_size, void* d_ws, size_t ws_size,
                              hipStream_t stream) {
  const float* x        = (const float*)d_in[0];
  const float* dl       = (const float*)d_in[1];
  const float* cpe_w    = (const float*)d_in[2];
  const float* cpe_b    = (const float*)d_in[3];
  const float* ln1_g    = (const float*)d_in[4];
  const float* ln1_b    = (const float*)d_in[5];
  const float* qkv_w    = (const float*)d_in[6];
  const float* proj_w   = (const float*)d_in[7];
  const float* proj_b   = (const float*)d_in[8];
  const float* dl_w1    = (const float*)d_in[9];
  const float* dl_b1    = (const float*)d_in[10];
  const float* dl_w2    = (const float*)d_in[11];
  const float* dl_b2    = (const float*)d_in[12];
  const float* crpe_w3  = (const float*)d_in[13];
  const float* crpe_b3  = (const float*)d_in[14];
  const float* crpe_w5  = (const float*)d_in[15];
  const float* crpe_b5  = (const float*)d_in[16];
  const float* crpe_w7  = (const float*)d_in[17];
  const float* crpe_b7  = (const float*)d_in[18];
  const float* ln2_g    = (const float*)d_in[19];
  const float* ln2_b    = (const float*)d_in[20];
  const float* fc1_w    = (const float*)d_in[21];
  const float* fc1_b    = (const float*)d_in[22];
  const float* fc2_w    = (const float*)d_in[23];
  const float* fc2_b    = (const float*)d_in[24];
  const int*   dptr     = (const int*)d_in[27];
  float* out = (float*)d_out;

  // ---- workspace carve-up ----
  char* p = (char*)d_ws;
  auto take = [&](size_t bytes) { char* r = p; p += (bytes + 255) & ~(size_t)255; return r; };
  float*  x1      = (float*) take((size_t)MM * Cc * 4);
  float*  qkv     = (float*) take((size_t)MM * 768 * 4);
  float*  x2      = (float*) take((size_t)MM * Cc * 4);
  float*  conv_v  = (float*) take((size_t)MM * Cc * 4);
  float*  kv      = (float*) take((size_t)Bn * NH * 1024 * 4);
  float*  da      = (float*) take((size_t)Bn * Cc * 4);
  bf16_t* cur_bf  = (bf16_t*)take((size_t)MM * Cc * 2);
  bf16_t* attn_bf = (bf16_t*)take((size_t)MM * Cc * 2);
  bf16_t* hid_bf  = (bf16_t*)take((size_t)MM * HID * 2);
  bf16_t* wq_bf   = (bf16_t*)take((size_t)768 * Cc * 2);
  bf16_t* wp_bf   = (bf16_t*)take((size_t)Cc * Cc * 2);
  bf16_t* w1_bf   = (bf16_t*)take((size_t)HID * Cc * 2);
  bf16_t* w2_bf   = (bf16_t*)take((size_t)Cc * HID * 2);

  // ---- weight conversion (transpose to [N,K] bf16) ----
  wconv_kernel<<<(Cc * 768 + 255) / 256, 256, 0, stream>>>(qkv_w, wq_bf, Cc, 768);
  wconv_kernel<<<(Cc * Cc + 255) / 256, 256, 0, stream>>>(proj_w, wp_bf, Cc, Cc);
  wconv_kernel<<<(Cc * HID + 255) / 256, 256, 0, stream>>>(fc1_w, w1_bf, Cc, HID);
  wconv_kernel<<<(HID * Cc + 255) / 256, 256, 0, stream>>>(fc2_w, w2_bf, HID, Cc);

  // ---- CPE (+residual) ----
  cpe_kernel<<<MM, 256, 0, stream>>>(x, cpe_w, cpe_b, x1);

  // ---- LN1 -> bf16 ----
  ln_kernel<<<MM, 256, 0, stream>>>(x1, ln1_g, ln1_b, dptr, cur_bf);

  // ---- qkv GEMM: [MM,256] x [256,768] ----
  {
    dim3 g(MM / 128, 768 / 64);
    gemm_bf16_kernel<EPI_PLAIN><<<g, 128, 0, stream>>>(
        cur_bf, wq_bf, nullptr, nullptr, qkv, nullptr, MM, 768, Cc);
  }

  // ---- token softmax + kv einsum ----
  kv_kernel<<<Bn * NH, 256, 0, stream>>>(qkv, kv);

  // ---- CRPE conv ----
  crpe_kernel<<<MM, 256, 0, stream>>>(qkv, crpe_w3, crpe_b3, crpe_w5, crpe_b5,
                                      crpe_w7, crpe_b7, conv_v);

  // ---- domain gating ----
  da_kernel<<<1, 256, 0, stream>>>(dl, dl_w1, dl_b1, dl_w2, dl_b2, da);

  // ---- fuse factor-att + crpe + gating -> bf16 attn ----
  fuse_kernel<<<MM, 256, 0, stream>>>(qkv, kv, conv_v, da, attn_bf);

  // ---- proj GEMM + bias + residual(x1) -> x2 ----
  {
    dim3 g(MM / 128, Cc / 64);
    gemm_bf16_kernel<EPI_BIAS_RESID><<<g, 128, 0, stream>>>(
        attn_bf, wp_bf, proj_b, x1, x2, nullptr, MM, Cc, Cc);
  }

  // ---- LN2 -> bf16 ----
  ln_kernel<<<MM, 256, 0, stream>>>(x2, ln2_g, ln2_b, dptr, cur_bf);

  // ---- fc1 GEMM + bias + exact GELU -> bf16 hidden ----
  {
    dim3 g(MM / 128, HID / 64);
    gemm_bf16_kernel<EPI_BIAS_GELU><<<g, 128, 0, stream>>>(
        cur_bf, w1_bf, fc1_b, nullptr, nullptr, hid_bf, MM, HID, Cc);
  }

  // ---- fc2 GEMM + bias + residual(x2) -> out ----
  {
    dim3 g(MM / 128, Cc / 64);
    gemm_bf16_kernel<EPI_BIAS_RESID><<<g, 128, 0, stream>>>(
        hid_bf, w2_bf, fc2_b, x2, out, nullptr, MM, Cc, HID);
  }
}
